// Block_75806172775137
// MI455X (gfx1250) — matrix operand order repaired and checked by reading, not compiled, not run on hardware
//
#include <hip/hip_runtime.h>

// ---------------------------------------------------------------------------
// Types
// ---------------------------------------------------------------------------
typedef _Float16 half_t;
typedef __attribute__((ext_vector_type(16))) _Float16 v16h;
typedef __attribute__((ext_vector_type(8)))  _Float16 v8h;
typedef __attribute__((ext_vector_type(8)))  float    v8f;

union Frag16 { v16h v; v8h h[2]; };
union Acc8   { v8f  v; float f[8]; };

// Problem constants (fixed by the reference)
#define Bsz 64
#define Ntok 197
#define Cdim 768
#define Hn 12
#define Dh 64
#define Hd 3072
#define KP 224         // padded K for p@v (197 -> 224, multiple of 32)
#define TOK (Bsz*Ntok) // 12608

// ---------------------------------------------------------------------------
// Generic WMMA GEMM:  C[M,Nout] = A[M,K] @ W[Nout,K]^T, batched over blockIdx.z
// Block = 256 threads (8 waves), tile 128x128, K-step 32, double-buffered LDS.
//
// OOB policy: tile row indices are CLAMPED (not zero-filled). A garbage/dup A
// row only affects the same (guarded-out) C row; a garbage W row only affects
// the same (guarded-out) C column — so clamping is safe and branch-free.
// ---------------------------------------------------------------------------
template <class Epi>
__global__ __launch_bounds__(256)
void gemm_wmma_kernel(const half_t* __restrict__ A, const half_t* __restrict__ W,
                      int M, int Nout, int K, int lda, int ldw,
                      long long strideA, long long strideW, Epi epi)
{
    constexpr int PITCH = 40; // halves per LDS row (80B): conflict-free b128 reads
    __shared__ __align__(16) half_t As[2][128 * PITCH];
    __shared__ __align__(16) half_t Bs[2][128 * PITCH];

    const int tid  = threadIdx.x;
    const int bz   = blockIdx.z;
    A += (long long)bz * strideA;
    W += (long long)bz * strideW;
    const int m0 = blockIdx.y * 128;
    const int n0 = blockIdx.x * 128;
    const int lane = tid & 31;
    const int wid  = tid >> 5;
    const int lo   = lane & 15;
    const int hi   = lane >> 4;
    const int wm   = wid & 1;   // 2 waves over M (64 rows each)
    const int wn   = wid >> 1;  // 4 waves over N (32 cols each)

    // Staging map: thread covers chunk ch of rows r0 and r0+64 (16B chunks).
    const int r0 = tid >> 2;
    const int ch = (tid & 3) * 8;
    const half_t* Ar0 = A + (long long)min(m0 + r0,      M - 1)    * lda + ch;
    const half_t* Ar1 = A + (long long)min(m0 + r0 + 64, M - 1)    * lda + ch;
    const half_t* Wr0 = W + (long long)min(n0 + r0,      Nout - 1) * ldw + ch;
    const half_t* Wr1 = W + (long long)min(n0 + r0 + 64, Nout - 1) * ldw + ch;
    const int st0 = r0 * PITCH + ch;
    const int st1 = (r0 + 64) * PITCH + ch;

    Acc8 acc[4][2];
#pragma unroll
    for (int i = 0; i < 4; ++i)
#pragma unroll
        for (int j = 0; j < 2; ++j)
#pragma unroll
            for (int e = 0; e < 8; ++e) acc[i][j].f[e] = 0.0f;

    auto compute = [&](int buf) {
        Frag16 a[4], b[2];
#pragma unroll
        for (int mt = 0; mt < 4; ++mt) {
            // A layout: lane half hi holds K chunks {hi*8..} and {hi*8+16..}
            const half_t* p = &As[buf][(wm * 64 + mt * 16 + lo) * PITCH + hi * 8];
            a[mt].h[0] = *(const v8h*)p;
            a[mt].h[1] = *(const v8h*)(p + 16);
        }
#pragma unroll
        for (int nt = 0; nt < 2; ++nt) {
            // B layout: lane = column, half hi holds K chunks {hi*8..} and {hi*8+16..}
            const half_t* p = &Bs[buf][(wn * 32 + nt * 16 + lo) * PITCH + hi * 8];
            b[nt].h[0] = *(const v8h*)p;
            b[nt].h[1] = *(const v8h*)(p + 16);
        }
#pragma unroll
        for (int mt = 0; mt < 4; ++mt)
#pragma unroll
            for (int nt = 0; nt < 2; ++nt)
                acc[mt][nt].v = __builtin_amdgcn_wmma_f32_16x16x32_f16(
                    false, a[mt].v, false, b[nt].v, (short)0, acc[mt][nt].v,
                    false, false);
    };

    // Prologue: stage K-tile 0 into buffer 0.
    {
        v8h a0 = *(const v8h*)Ar0;
        v8h a1 = *(const v8h*)Ar1;
        v8h w0 = *(const v8h*)Wr0;
        v8h w1 = *(const v8h*)Wr1;
        *(v8h*)(&As[0][st0]) = a0;
        *(v8h*)(&As[0][st1]) = a1;
        *(v8h*)(&Bs[0][st0]) = w0;
        *(v8h*)(&Bs[0][st1]) = w1;
    }
    __syncthreads();

    int buf = 0;
    int k0 = 0;
    for (; k0 + 32 < K; k0 += 32, buf ^= 1) {
        // Issue next-tile global loads (overlap with WMMA burst below).
        v8h a0 = *(const v8h*)(Ar0 + k0 + 32);
        v8h a1 = *(const v8h*)(Ar1 + k0 + 32);
        v8h w0 = *(const v8h*)(Wr0 + k0 + 32);
        v8h w1 = *(const v8h*)(Wr1 + k0 + 32);

        compute(buf);

        *(v8h*)(&As[buf ^ 1][st0]) = a0;
        *(v8h*)(&As[buf ^ 1][st1]) = a1;
        *(v8h*)(&Bs[buf ^ 1][st0]) = w0;
        *(v8h*)(&Bs[buf ^ 1][st1]) = w1;
        __syncthreads();
    }
    compute(buf); // last K-tile

    // Epilogue (boundary-guarded stores).
#pragma unroll
    for (int mt = 0; mt < 4; ++mt)
#pragma unroll
        for (int nt = 0; nt < 2; ++nt) {
            int col = n0 + wn * 32 + nt * 16 + lo;
            if (col >= Nout) continue;
#pragma unroll
            for (int j = 0; j < 8; ++j) {
                int row = m0 + wm * 64 + mt * 16 + hi * 8 + j;
                if (row < M) epi(bz, row, col, acc[mt][nt].f[j]);
            }
        }
}

// ---------------------------------------------------------------------------
// Epilogues
// ---------------------------------------------------------------------------
struct EpiQKV {
    const float* bias; half_t* q; half_t* k; half_t* vt;
    __device__ void operator()(int, int row, int col, float acc) const {
        float v = acc + bias[col];
        int which = col / Cdim, c = col - which * Cdim;
        int h = c >> 6, d = c & 63;
        int b = row / Ntok, n = row - b * Ntok;
        int bh = b * Hn + h;
        if (which == 0)      q[((long long)bh * Ntok + n) * Dh + d] = (half_t)(v * 0.125f);
        else if (which == 1) k[((long long)bh * Ntok + n) * Dh + d] = (half_t)v;
        else                 vt[((long long)bh * Dh + d) * KP + n]  = (half_t)v;
    }
};

struct EpiScores {
    float* attn;
    __device__ void operator()(int bz, int row, int col, float acc) const {
        attn[((long long)bz * Ntok + row) * Ntok + col] = acc;
    }
};

struct EpiPV {
    half_t* o;
    __device__ void operator()(int bz, int row, int col, float acc) const {
        int b = bz / Hn, h = bz - b * Hn;
        o[((long long)(b * Ntok + row)) * Cdim + h * Dh + col] = (half_t)acc;
    }
};

struct EpiProj {
    const float* bias; const float* x; const float* keep; float* x1;
    __device__ void operator()(int, int row, int col, float acc) const {
        float v = acc + bias[col];
        int b = row / Ntok, n = row - b * Ntok;
        float km = keep[b * Ntok + n];
        long long idx = (long long)row * Cdim + col;
        x1[idx] = x[idx] + v * km;
    }
};

struct EpiFC1 {
    const float* bias; half_t* m1;
    __device__ void operator()(int, int row, int col, float acc) const {
        float v = acc + bias[col];
        float g = 0.5f * v * (1.0f + erff(v * 0.70710678118654752f)); // exact GELU
        m1[(long long)row * Hd + col] = (half_t)g;
    }
};

struct EpiFC2 {
    const float* bias; const float* x1; const float* keep; float* xout;
    __device__ void operator()(int, int row, int col, float acc) const {
        float v = acc + bias[col];
        int b = row / Ntok, n = row - b * Ntok;
        float km = keep[b * Ntok + n];
        long long idx = (long long)row * Cdim + col;
        xout[idx] = x1[idx] + v * km;
    }
};

// ---------------------------------------------------------------------------
// Elementwise / reduction kernels
// ---------------------------------------------------------------------------
__global__ void f32_to_f16_kernel(const float* __restrict__ in, half_t* __restrict__ out,
                                  long long n) {
    for (long long i = (long long)blockIdx.x * blockDim.x + threadIdx.x; i < n;
         i += (long long)gridDim.x * blockDim.x)
        out[i] = (half_t)in[i];
}

__global__ void fill_f16_zero_kernel(half_t* __restrict__ p, long long n) {
    for (long long i = (long long)blockIdx.x * blockDim.x + threadIdx.x; i < n;
         i += (long long)gridDim.x * blockDim.x)
        p[i] = (half_t)0.0f;
}

// One block per token row; LayerNorm over 768, fp16 output.
__global__ __launch_bounds__(256)
void ln_f16_kernel(const float* __restrict__ x, const float* __restrict__ g,
                   const float* __restrict__ b, half_t* __restrict__ out) {
    long long t = blockIdx.x;
    const float* row = x + t * Cdim;
    int i = threadIdx.x;
    float v0 = row[i], v1 = row[i + 256], v2 = row[i + 512];
    __shared__ float red[256];
    red[i] = v0 + v1 + v2;
    __syncthreads();
    for (int off = 128; off > 0; off >>= 1) {
        if (i < off) red[i] += red[i + off];
        __syncthreads();
    }
    float mu = red[0] * (1.0f / Cdim);
    __syncthreads();
    float d0 = v0 - mu, d1 = v1 - mu, d2 = v2 - mu;
    red[i] = d0 * d0 + d1 * d1 + d2 * d2;
    __syncthreads();
    for (int off = 128; off > 0; off >>= 1) {
        if (i < off) red[i] += red[i + off];
        __syncthreads();
    }
    float rstd = rsqrtf(red[0] * (1.0f / Cdim) + 1e-5f);
    half_t* orow = out + t * Cdim;
    orow[i]       = (half_t)(d0 * rstd * g[i]       + b[i]);
    orow[i + 256] = (half_t)(d1 * rstd * g[i + 256] + b[i + 256]);
    orow[i + 512] = (half_t)(d2 * rstd * g[i + 512] + b[i + 512]);
}

// cls_scores[b][j] = mean_h attn[b,h,0,j+1], j in [0,196)
__global__ void cls_mean_kernel(const float* __restrict__ attn, float* __restrict__ cls) {
    int b = blockIdx.x, j = threadIdx.x;
    if (j < Ntok - 1) {
        float s = 0.0f;
        for (int h = 0; h < Hn; ++h)
            s += attn[((long long)(b * Hn + h) * Ntok) * Ntok + (j + 1)];
        cls[b * (Ntok - 1) + j] = s * (1.0f / Hn);
    }
}

// Exact rank-based top-k (ties broken by lower index, matching jax.lax.top_k).
__global__ void topk_keep_kernel(const float* __restrict__ cls, const int* __restrict__ nk_p,
                                 float* __restrict__ keep, float* __restrict__ keep_out) {
    int b = blockIdx.x, j = threadIdx.x;
    __shared__ float s[Ntok - 1];
    if (j < Ntok - 1) s[j] = cls[b * (Ntok - 1) + j];
    __syncthreads();
    int nk = *nk_p;
    if (j < Ntok - 1) {
        float sj = s[j];
        int rank = 0;
        for (int i = 0; i < Ntok - 1; ++i) {
            float si = s[i];
            rank += (si > sj) || (si == sj && i < j);
        }
        float kv = (rank < nk) ? 1.0f : 0.0f;
        keep[b * Ntok + j + 1]     = kv;
        keep_out[b * Ntok + j + 1] = kv;
    }
    if (j == 0) { keep[b * Ntok] = 1.0f; keep_out[b * Ntok] = 1.0f; }
}

// Masked softmax over one attention row -> fp16 p, row padded to KP with zeros.
__global__ __launch_bounds__(256)
void softmax_kernel(const float* __restrict__ attn, const float* __restrict__ keep,
                    half_t* __restrict__ p) {
    int blk = blockIdx.x;
    int bh = blk / Ntok, row = blk - bh * Ntok;
    int b = bh / Hn;
    const float* arow = attn + ((long long)bh * Ntok + row) * Ntok;
    const float* krow = keep + b * Ntok;
    int t = threadIdx.x;
    float val = -3.0e38f;
    if (t < Ntok) val = arow[t] + (1.0f - krow[t]) * (-100000.0f);
    __shared__ float red[256];
    red[t] = val;
    __syncthreads();
    for (int off = 128; off > 0; off >>= 1) {
        if (t < off) red[t] = fmaxf(red[t], red[t + off]);
        __syncthreads();
    }
    float mx = red[0];
    __syncthreads();
    float e = (t < Ntok) ? __expf(val - mx) : 0.0f;
    red[t] = e;
    __syncthreads();
    for (int off = 128; off > 0; off >>= 1) {
        if (t < off) red[t] += red[t + off];
        __syncthreads();
    }
    float inv = 1.0f / red[0];
    half_t* prow = p + ((long long)bh * Ntok + row) * KP;
    if (t < KP) prow[t] = (t < Ntok) ? (half_t)(e * inv) : (half_t)0.0f;
}

// ---------------------------------------------------------------------------
// Workspace layout (bytes, all 256-aligned). m1 aliases dead q/k/vt/p; h2 aliases h.
// ---------------------------------------------------------------------------
static constexpr size_t OFF_H    = 0;                               // TOK*C f16
static constexpr size_t OFF_Q    = OFF_H    + (size_t)TOK * Cdim * 2;
static constexpr size_t OFF_K    = OFF_Q    + (size_t)TOK * Cdim * 2;
static constexpr size_t OFF_VT   = OFF_K    + (size_t)TOK * Cdim * 2;          // B*H*Dh*KP f16
static constexpr size_t OFF_P    = OFF_VT   + (size_t)Bsz * Hn * Dh * KP * 2;  // B*H*N*KP f16
static constexpr size_t OFF_O    = OFF_P    + (size_t)Bsz * Hn * Ntok * KP * 2;
static constexpr size_t OFF_X1   = OFF_O    + (size_t)TOK * Cdim * 2;          // f32
static constexpr size_t OFF_WQ   = OFF_X1   + (size_t)TOK * Cdim * 4;
static constexpr size_t OFF_WP   = OFF_WQ   + (size_t)3 * Cdim * Cdim * 2;
static constexpr size_t OFF_W1   = OFF_WP   + (size_t)Cdim * Cdim * 2;
static constexpr size_t OFF_W2   = OFF_W1   + (size_t)Hd * Cdim * 2;
static constexpr size_t OFF_CLS  = OFF_W2   + (size_t)Cdim * Hd * 2;
static constexpr size_t OFF_KEEP = OFF_CLS  + (size_t)Bsz * (Ntok - 1) * 4 + 256;
static constexpr size_t OFF_M1   = OFF_Q;   // alias: q/k/vt/p dead by FC1 time
static constexpr size_t OFF_H2   = OFF_H;   // alias: h dead after QKV

// ---------------------------------------------------------------------------
extern "C" void kernel_launch(void* const* d_in, const int* in_sizes, int n_in,
                              void* d_out, int out_size, void* d_ws, size_t ws_size,
                              hipStream_t stream) {
    const float* x      = (const float*)d_in[0];
    const float* ln1_g  = (const float*)d_in[1];
    const float* ln1_b  = (const float*)d_in[2];
    const float* qkv_w  = (const float*)d_in[3];
    const float* qkv_b  = (const float*)d_in[4];
    const float* proj_w = (const float*)d_in[5];
    const float* proj_b = (const float*)d_in[6];
    const float* ln2_g  = (const float*)d_in[7];
    const float* ln2_b  = (const float*)d_in[8];
    const float* fc1_w  = (const float*)d_in[9];
    const float* fc1_b  = (const float*)d_in[10];
    const float* fc2_w  = (const float*)d_in[11];
    const float* fc2_b  = (const float*)d_in[12];
    const int*   nkeep  = (const int*)d_in[13];

    float* out      = (float*)d_out;
    float* x_out    = out;
    float* keep_out = out + (size_t)TOK * Cdim;
    float* attn_out = out + (size_t)TOK * Cdim + TOK;

    char* ws = (char*)d_ws;
    half_t* h   = (half_t*)(ws + OFF_H);
    half_t* q   = (half_t*)(ws + OFF_Q);
    half_t* k   = (half_t*)(ws + OFF_K);
    half_t* vt  = (half_t*)(ws + OFF_VT);
    half_t* p   = (half_t*)(ws + OFF_P);
    half_t* o   = (half_t*)(ws + OFF_O);
    float*  x1  = (float*)(ws + OFF_X1);
    half_t* wq  = (half_t*)(ws + OFF_WQ);
    half_t* wp  = (half_t*)(ws + OFF_WP);
    half_t* w1  = (half_t*)(ws + OFF_W1);
    half_t* w2  = (half_t*)(ws + OFF_W2);
    float*  cls = (float*)(ws + OFF_CLS);
    float*  kp  = (float*)(ws + OFF_KEEP);
    half_t* m1  = (half_t*)(ws + OFF_M1);
    half_t* h2  = (half_t*)(ws + OFF_H2);

    // 1) fp16 weight copies
    f32_to_f16_kernel<<<4096, 256, 0, stream>>>(qkv_w, wq, (long long)3 * Cdim * Cdim);
    f32_to_f16_kernel<<<2048, 256, 0, stream>>>(proj_w, wp, (long long)Cdim * Cdim);
    f32_to_f16_kernel<<<4096, 256, 0, stream>>>(fc1_w, w1, (long long)Hd * Cdim);
    f32_to_f16_kernel<<<4096, 256, 0, stream>>>(fc2_w, w2, (long long)Cdim * Hd);
    // zero v^T (padding columns 197..223 must be 0 for the p@v GEMM)
    fill_f16_zero_kernel<<<4096, 256, 0, stream>>>(vt, (long long)Bsz * Hn * Dh * KP);

    // 2) LN1 -> h (f16)
    ln_f16_kernel<<<TOK, 256, 0, stream>>>(x, ln1_g, ln1_b, h);

    // 3) QKV GEMM: (12608,768)x(2304,768)^T, scatter q/k/v^T
    {
        EpiQKV epi{qkv_b, q, k, vt};
        dim3 g((3 * Cdim + 127) / 128, (TOK + 127) / 128, 1);
        gemm_wmma_kernel<EpiQKV><<<g, 256, 0, stream>>>(h, wq, TOK, 3 * Cdim, Cdim,
                                                        Cdim, Cdim, 0, 0, epi);
    }
    // 4) Scores: per (b,h) (197,64)x(197,64)^T -> attn_rt (d_out)
    {
        EpiScores epi{attn_out};
        dim3 g((Ntok + 127) / 128, (Ntok + 127) / 128, Bsz * Hn);
        gemm_wmma_kernel<EpiScores><<<g, 256, 0, stream>>>(q, k, Ntok, Ntok, Dh,
                                                           Dh, Dh,
                                                           (long long)Ntok * Dh,
                                                           (long long)Ntok * Dh, epi);
    }
    // 5) CLS mean + top-k keep mask
    cls_mean_kernel<<<Bsz, 256, 0, stream>>>(attn_out, cls);
    topk_keep_kernel<<<Bsz, 256, 0, stream>>>(cls, nkeep, kp, keep_out);
    // 6) masked softmax -> p (f16, KP-padded)
    softmax_kernel<<<Bsz * Hn * Ntok, 256, 0, stream>>>(attn_out, kp, p);
    // 7) o = p @ v : per (b,h) (197,224)x(64,224)^T
    {
        EpiPV epi{o};
        dim3 g(1, (Ntok + 127) / 128, Bsz * Hn);
        gemm_wmma_kernel<EpiPV><<<g, 256, 0, stream>>>(p, vt, Ntok, Dh, KP,
                                                       KP, KP,
                                                       (long long)Ntok * KP,
                                                       (long long)Dh * KP, epi);
    }
    // 8) proj GEMM + residual*keep -> x1 (f32)
    {
        EpiProj epi{proj_b, x, kp, x1};
        dim3 g((Cdim + 127) / 128, (TOK + 127) / 128, 1);
        gemm_wmma_kernel<EpiProj><<<g, 256, 0, stream>>>(o, wp, TOK, Cdim, Cdim,
                                                         Cdim, Cdim, 0, 0, epi);
    }
    // 9) LN2 -> h2 (f16, aliases h)
    ln_f16_kernel<<<TOK, 256, 0, stream>>>(x1, ln2_g, ln2_b, h2);
    // 10) FC1 + exact GELU -> m1 (f16)
    {
        EpiFC1 epi{fc1_b, m1};
        dim3 g((Hd + 127) / 128, (TOK + 127) / 128, 1);
        gemm_wmma_kernel<EpiFC1><<<g, 256, 0, stream>>>(h2, w1, TOK, Hd, Cdim,
                                                        Cdim, Cdim, 0, 0, epi);
    }
    // 11) FC2 + residual*keep -> x_out (d_out)
    {
        EpiFC2 epi{fc2_b, x1, kp, x_out};
        dim3 g((Cdim + 127) / 128, (TOK + 127) / 128, 1);
        gemm_wmma_kernel<EpiFC2><<<g, 256, 0, stream>>>(m1, w2, TOK, Cdim, Hd,
                                                        Hd, Hd, 0, 0, epi);
    }
    (void)n_in; (void)in_sizes; (void)out_size; (void)ws_size;
}
